// MultiHeadSelfAttention_16793322127913
// MI455X (gfx1250) — matrix operand order repaired and checked by reading, not compiled, not run on hardware
//
#include <hip/hip_runtime.h>

typedef _Float16 half8 __attribute__((ext_vector_type(8)));
typedef _Float16 v16h  __attribute__((ext_vector_type(16)));
typedef float    v8f   __attribute__((ext_vector_type(8)));
typedef unsigned int u32x4 __attribute__((ext_vector_type(4)));
typedef int          i32x4 __attribute__((ext_vector_type(4)));
typedef int          i32x8 __attribute__((ext_vector_type(8)));

#define DM    1024
#define HEADS 16
#define DK    64
#define SEQ   2048
#define BATCH 2
#define MTOT  (BATCH * SEQ)   // 4096 token rows

// ---------------------------------------------------------------------------
// WMMA helpers (v_wmma_f32_16x16x32_f16): D = A(16x32 f16) * B(32x16 f16) + C
// ---------------------------------------------------------------------------
__device__ __forceinline__ v8f wmma_f16(v16h a, v16h b, v8f c) {
  return __builtin_amdgcn_wmma_f32_16x16x32_f16(
      /*neg_a=*/false, a, /*neg_b=*/false, b,
      /*c_mod=*/(short)0, c, /*reuse_a=*/false, /*reuse_b=*/false);
}

// A fragment: 16x32 (MxK), row-major source with leading dim ld.
__device__ __forceinline__ v16h load_frag_a(const _Float16* __restrict__ p,
                                            int ld, int row0, int k0, int lane) {
  const int r    = row0 + (lane & 15);
  const int koff = k0 + ((lane >> 4) << 3);   // +8 for upper half-wave
  const _Float16* q = p + (size_t)r * ld + koff;
  half8 lo = *(const half8*)(q);
  half8 hi = *(const half8*)(q + 16);
  v16h out;
#pragma unroll
  for (int i = 0; i < 8; ++i) { out[i] = lo[i]; out[8 + i] = hi[i]; }
  return out;
}

// B fragment: 32x16 (KxN). Source stored N-major (row n contiguous in K).
__device__ __forceinline__ v16h load_frag_b(const _Float16* __restrict__ p,
                                            int ld, int n0, int k0, int lane) {
  const int n    = n0 + (lane & 15);
  const int koff = k0 + ((lane >> 4) << 3);   // +8 for upper half-wave
  const _Float16* q = p + (size_t)n * ld + koff;
  half8 lo = *(const half8*)(q);
  half8 hi = *(const half8*)(q + 16);
  v16h out;
#pragma unroll
  for (int i = 0; i < 8; ++i) { out[i] = lo[i]; out[8 + i] = hi[i]; }
  return out;
}

// ---------------------------------------------------------------------------
// Tensor Data Mover: 2D tile load Global -> LDS, element = 4 bytes (2 x f16).
// D# per CDNA5 ISA; amdgpu-toolchain 6-arg builtin.
// ---------------------------------------------------------------------------
__device__ __forceinline__ void tdm_load_2d(const void* gptr, void* ldsptr,
                                            unsigned tensor_d0_dw,  // row len (dwords)
                                            unsigned tensor_d1,     // #rows
                                            unsigned stride_dw,     // row stride (dwords)
                                            unsigned tile_d0_dw,    // tile row len (dwords)
                                            unsigned tile_d1) {     // tile #rows
  const unsigned long long ga = (unsigned long long)(size_t)gptr;
  const unsigned lds_off = (unsigned)(size_t)ldsptr;   // generic low 32b = LDS offset
  u32x4 g0;
  g0[0] = 1u;                                          // count=1, user descriptor
  g0[1] = lds_off;                                     // bits 63:32 lds_addr
  g0[2] = (unsigned)ga;                                // global_addr[31:0]
  g0[3] = (unsigned)((ga >> 32) & 0x01ffffffull) | (2u << 30);  // addr[56:32] | type=2
  i32x8 g1;
  g1[0] = (int)(2u << 16);                             // workgroup_mask=0, data_size=2 (4B)
  g1[1] = (int)((tensor_d0_dw & 0xffffu) << 16);       // tensor_dim0[15:0]
  g1[2] = (int)((tensor_d0_dw >> 16) | ((tensor_d1 & 0xffffu) << 16));
  g1[3] = (int)((tensor_d1 >> 16) | ((tile_d0_dw & 0xffffu) << 16));
  g1[4] = (int)(tile_d1 & 0xffffu);                    // tile_dim1, tile_dim2=0
  g1[5] = (int)stride_dw;                              // tensor_dim0_stride[31:0]
  g1[6] = 0;                                           // stride hi / dim1_stride lo
  g1[7] = 0;
  const i32x4 z4 = {0, 0, 0, 0};                       // groups 2/3 unused (2D)
  const i32x8 z8 = {0, 0, 0, 0, 0, 0, 0, 0};
  __builtin_amdgcn_tensor_load_to_lds(g0, g1, z4, z4, z8, 0);
}

// ---------------------------------------------------------------------------
// Stage 1: fp32 -> f16 conversion
// ---------------------------------------------------------------------------
__global__ void cvt_f32_f16(const float* __restrict__ in,
                            _Float16* __restrict__ out, int n) {
  for (int i = blockIdx.x * blockDim.x + threadIdx.x; i < n;
       i += gridDim.x * blockDim.x)
    out[i] = (_Float16)in[i];
}

// ---------------------------------------------------------------------------
// Stage 2: QKV projection + RoPE. One wave computes a 64x16 tile of x @ W^T.
// which = blockIdx.z: 0->Q (rope), 1->K (rope), 2->V (stored transposed [d][s])
// ---------------------------------------------------------------------------
__global__ void qkv_rope_kernel(const _Float16* __restrict__ x16,
                                const _Float16* __restrict__ Wq16,
                                const _Float16* __restrict__ Wk16,
                                const _Float16* __restrict__ Wv16,
                                const float* __restrict__ cosT,
                                const float* __restrict__ sinT,
                                const int*   __restrict__ tokpos,
                                _Float16* __restrict__ Qh,
                                _Float16* __restrict__ Kh,
                                _Float16* __restrict__ Vt) {
  const int lane  = threadIdx.x;
  const int m0    = blockIdx.x * 64;
  const int n0    = blockIdx.y * 16;
  const int which = blockIdx.z;
  const _Float16* W = (which == 0) ? Wq16 : (which == 1) ? Wk16 : Wv16;

  v8f acc[4];
#pragma unroll
  for (int t = 0; t < 4; ++t)
#pragma unroll
    for (int i = 0; i < 8; ++i) acc[t][i] = 0.0f;

  for (int k0 = 0; k0 < DM; k0 += 32) {
    v16h b = load_frag_b(W, DM, n0, k0, lane);
#pragma unroll
    for (int t = 0; t < 4; ++t) {
      v16h a = load_frag_a(x16, DM, m0 + t * 16, k0, lane);
      acc[t] = wmma_f16(a, b, acc[t]);
    }
  }

  const int n    = lane & 15;
  const int off  = (lane >> 4) << 3;
  const int ncol = n0 + n;
  const int h    = ncol >> 6;
  const int d    = ncol & 63;

#pragma unroll
  for (int t = 0; t < 4; ++t) {
    const int rowbase = m0 + t * 16 + off;
    const int bidx    = rowbase >> 11;           // / SEQ
    const int sbase   = rowbase & (SEQ - 1);
    if (which == 2) {
      half8 v;
#pragma unroll
      for (int i = 0; i < 8; ++i) v[i] = (_Float16)acc[t][i];
      size_t idx = (((size_t)(bidx * HEADS + h)) * DK + d) * SEQ + sbase;
      *(half8*)(Vt + idx) = v;
    } else {
      _Float16* dst = (which == 0) ? Qh : Kh;
#pragma unroll
      for (int i = 0; i < 8; ++i) {
        const int s   = sbase + i;
        const int pos = tokpos[s];
        const float c  = cosT[pos * DK + d];
        const float sn = sinT[pos * DK + d];
        const float v  = acc[t][i];
        const float pr = __shfl_xor(v, 1);       // partner dim d^1 = lane^1
        const float rot = (d & 1) ? pr : -pr;
        const float r   = v * c + rot * sn;
        size_t idx = (((size_t)(bidx * HEADS + h)) * SEQ + s) * DK + d;
        dst[idx] = (_Float16)r;
      }
    }
  }
}

// ---------------------------------------------------------------------------
// Stage 3: causal flash attention. 4 waves / block, 64 queries per block.
// K/V key-tiles staged into double-buffered LDS by the Tensor Data Mover,
// DMA of tile kb+1 overlapped with compute on tile kb (s_wait_tensorcnt).
// ---------------------------------------------------------------------------
__global__ void attn_kernel(const _Float16* __restrict__ Qh,
                            const _Float16* __restrict__ Kh,
                            const _Float16* __restrict__ Vt,
                            _Float16* __restrict__ O16) {
  __shared__ __align__(16) _Float16 ldsK[2][32 * DK];   // 2 x (32 keys x 64 d)
  __shared__ __align__(16) _Float16 ldsV[2][DK * 32];   // 2 x (64 d x 32 s)
  __shared__ __align__(16) _Float16 pbuf[4][16][32];    // per-wave P bounce

  const int tid   = threadIdx.x;
  const int lane  = tid & 31;
  const int w     = tid >> 5;
  const int qb    = blockIdx.x;                 // 64-query block
  const int bh    = blockIdx.y;
  const int qrow0 = qb * 64 + w * 16;

  const _Float16* Q = Qh + (size_t)bh * SEQ * DK;
  const _Float16* K = Kh + (size_t)bh * SEQ * DK;
  const _Float16* V = Vt + (size_t)bh * DK * SEQ;

  const v16h qa0 = load_frag_a(Q, DK, qrow0, 0, lane);
  const v16h qa1 = load_frag_a(Q, DK, qrow0, 32, lane);

  v8f acc[4];
  float m_i[8], l_i[8];
#pragma unroll
  for (int i = 0; i < 8; ++i) { m_i[i] = -__builtin_inff(); l_i[i] = 0.0f; }
#pragma unroll
  for (int t = 0; t < 4; ++t)
#pragma unroll
    for (int i = 0; i < 8; ++i) acc[t][i] = 0.0f;

  const int n   = lane & 15;
  const int off = (lane >> 4) << 3;
  const int nkb = 2 * qb + 2;                   // causal: keys up to qb*64+63

  // Prime the pipeline: DMA tile 0 into buffer 0 (wave 0 only; TDM ignores EXEC).
  if (w == 0) {
    tdm_load_2d(K, ldsK[0], DK / 2, 32, DK / 2, DK / 2, 32);
    tdm_load_2d(V, ldsV[0], SEQ / 2, DK, SEQ / 2, 16, DK);
  }

  for (int kb = 0; kb < nkb; ++kb) {
    const int kbase = kb * 32;
    const int cur   = kb & 1;

    if (w == 0) {
      if (kb + 1 < nkb) {
        const int kn = kbase + 32;
        tdm_load_2d(K + (size_t)kn * DK, ldsK[cur ^ 1], DK / 2, 32, DK / 2, DK / 2, 32);
        tdm_load_2d(V + kn,              ldsV[cur ^ 1], SEQ / 2, DK, SEQ / 2, 16, DK);
        __builtin_amdgcn_s_wait_tensorcnt(2);   // tile kb landed (in-order TDM)
      } else {
        __builtin_amdgcn_s_wait_tensorcnt(0);
      }
    }
    __syncthreads();                            // release compute waves

    if (kbase <= qrow0 + 15) {                  // wave-uniform causal skip
      const _Float16* lk = ldsK[cur];
      const _Float16* lv = ldsV[cur];

      v8f c0, c1;
#pragma unroll
      for (int i = 0; i < 8; ++i) { c0[i] = 0.0f; c1[i] = 0.0f; }
      c0 = wmma_f16(qa0, load_frag_b(lk, DK, 0,  0,  lane), c0);
      c0 = wmma_f16(qa1, load_frag_b(lk, DK, 0,  32, lane), c0);
      c1 = wmma_f16(qa0, load_frag_b(lk, DK, 16, 0,  lane), c1);
      c1 = wmma_f16(qa1, load_frag_b(lk, DK, 16, 32, lane), c1);

#pragma unroll
      for (int i = 0; i < 8; ++i) {
        const int rowpos = qrow0 + off + i;
        float s0 = c0[i] * 0.125f;              // 1/sqrt(64)
        float s1 = c1[i] * 0.125f;
        if (kbase + n      > rowpos) s0 = -__builtin_inff();
        if (kbase + 16 + n > rowpos) s1 = -__builtin_inff();

        float t = fmaxf(s0, s1);
#pragma unroll
        for (int o = 1; o < 16; o <<= 1) t = fmaxf(t, __shfl_xor(t, o));
        const float mnew  = fmaxf(m_i[i], t);
        const float alpha = __expf(m_i[i] - mnew);
        const float p0 = __expf(s0 - mnew);
        const float p1 = __expf(s1 - mnew);
        float rs = p0 + p1;
#pragma unroll
        for (int o = 1; o < 16; o <<= 1) rs += __shfl_xor(rs, o);
        l_i[i] = l_i[i] * alpha + rs;
        m_i[i] = mnew;
#pragma unroll
        for (int t4 = 0; t4 < 4; ++t4) acc[t4][i] *= alpha;

        pbuf[w][off + i][n]      = (_Float16)p0;
        pbuf[w][off + i][16 + n] = (_Float16)p1;
      }

      // Reload P in A-fragment layout (wave-private LDS; DS is in-order per wave)
      v16h pa;
      {
        const _Float16* q = &pbuf[w][lane & 15][off];
        half8 lo = *(const half8*)(q);
        half8 hi = *(const half8*)(q + 16);
#pragma unroll
        for (int i = 0; i < 8; ++i) { pa[i] = lo[i]; pa[8 + i] = hi[i]; }
      }

      // O += P @ V  (Vt tile rows are d, 32 s contiguous)
#pragma unroll
      for (int td = 0; td < 4; ++td) {
        v16h vb = load_frag_b(lv, 32, td * 16, 0, lane);
        acc[td] = wmma_f16(pa, vb, acc[td]);
      }
    }
    __syncthreads();                            // reads done before next DMA lands
  }

  // Epilogue: normalize and store concatenated heads: O16[b][s][h*64+d]
  const int b = bh >> 4, h = bh & 15;
#pragma unroll
  for (int i = 0; i < 8; ++i) {
    const float inv = 1.0f / l_i[i];
    const int s = qrow0 + off + i;
#pragma unroll
    for (int td = 0; td < 4; ++td) {
      const float v = acc[td][i] * inv;
      size_t idx = ((size_t)(b * SEQ + s)) * DM + h * DK + td * 16 + n;
      O16[idx] = (_Float16)v;
    }
  }
}

// ---------------------------------------------------------------------------
// Stage 4: output projection out = O @ Wo^T (fp32 result)
// ---------------------------------------------------------------------------
__global__ void proj_kernel(const _Float16* __restrict__ O16,
                            const _Float16* __restrict__ Wo16,
                            float* __restrict__ out) {
  const int lane = threadIdx.x;
  const int m0 = blockIdx.x * 64;
  const int n0 = blockIdx.y * 16;

  v8f acc[4];
#pragma unroll
  for (int t = 0; t < 4; ++t)
#pragma unroll
    for (int i = 0; i < 8; ++i) acc[t][i] = 0.0f;

  for (int k0 = 0; k0 < DM; k0 += 32) {
    v16h b = load_frag_b(Wo16, DM, n0, k0, lane);
#pragma unroll
    for (int t = 0; t < 4; ++t) {
      v16h a = load_frag_a(O16, DM, m0 + t * 16, k0, lane);
      acc[t] = wmma_f16(a, b, acc[t]);
    }
  }

  const int n   = lane & 15;
  const int off = (lane >> 4) << 3;
#pragma unroll
  for (int t = 0; t < 4; ++t) {
    const int rowbase = m0 + t * 16 + off;
#pragma unroll
    for (int i = 0; i < 8; ++i)
      out[(size_t)(rowbase + i) * DM + n0 + n] = acc[t][i];
  }
}

// ---------------------------------------------------------------------------
// Host launcher
// ---------------------------------------------------------------------------
extern "C" void kernel_launch(void* const* d_in, const int* in_sizes, int n_in,
                              void* d_out, int out_size, void* d_ws, size_t ws_size,
                              hipStream_t stream) {
  const float* x    = (const float*)d_in[0];
  const float* Wq   = (const float*)d_in[1];
  const float* Wk   = (const float*)d_in[2];
  const float* Wv   = (const float*)d_in[3];
  const float* Wo   = (const float*)d_in[4];
  const float* cosT = (const float*)d_in[5];
  const float* sinT = (const float*)d_in[6];
  const int*   tok  = (const int*)d_in[7];
  float* out = (float*)d_out;

  char* ws = (char*)d_ws;
  const size_t SZ_X = (size_t)MTOT * DM * sizeof(_Float16);      // 8 MiB
  const size_t SZ_W = (size_t)DM * DM * sizeof(_Float16);        // 2 MiB
  const size_t SZ_T = (size_t)BATCH * HEADS * SEQ * DK * sizeof(_Float16); // 8 MiB
  _Float16* x16  = (_Float16*)(ws);
  _Float16* Wq16 = (_Float16*)(ws + SZ_X);
  _Float16* Wk16 = (_Float16*)(ws + SZ_X + SZ_W);
  _Float16* Wv16 = (_Float16*)(ws + SZ_X + 2 * SZ_W);
  _Float16* Wo16 = (_Float16*)(ws + SZ_X + 3 * SZ_W);
  _Float16* Qh   = (_Float16*)(ws + SZ_X + 4 * SZ_W);
  _Float16* Kh   = (_Float16*)(ws + SZ_X + 4 * SZ_W + SZ_T);
  _Float16* Vt   = (_Float16*)(ws + SZ_X + 4 * SZ_W + 2 * SZ_T);
  _Float16* O16  = (_Float16*)(ws + SZ_X + 4 * SZ_W + 3 * SZ_T);

  cvt_f32_f16<<<2048, 256, 0, stream>>>(x,  x16,  MTOT * DM);
  cvt_f32_f16<<<1024, 256, 0, stream>>>(Wq, Wq16, DM * DM);
  cvt_f32_f16<<<1024, 256, 0, stream>>>(Wk, Wk16, DM * DM);
  cvt_f32_f16<<<1024, 256, 0, stream>>>(Wv, Wv16, DM * DM);
  cvt_f32_f16<<<1024, 256, 0, stream>>>(Wo, Wo16, DM * DM);

  qkv_rope_kernel<<<dim3(MTOT / 64, DM / 16, 3), 32, 0, stream>>>(
      x16, Wq16, Wk16, Wv16, cosT, sinT, tok, Qh, Kh, Vt);

  attn_kernel<<<dim3(SEQ / 64, BATCH * HEADS), 128, 0, stream>>>(Qh, Kh, Vt, O16);

  proj_kernel<<<dim3(MTOT / 64, DM / 16), 32, 0, stream>>>(O16, Wo16, out);
}
